// Network_61564061221125
// MI455X (gfx1250) — matrix operand order repaired and checked by reading, not compiled, not run on hardware
//
#include <hip/hip_runtime.h>
#include <hip/hip_bf16.h>
#include <stdint.h>

typedef __attribute__((ext_vector_type(16))) _Float16 v16h;
typedef __attribute__((ext_vector_type(8)))  _Float16 half8;
typedef __attribute__((ext_vector_type(8)))  float    v8f;

#define R_RAYS 4096
#define NBOX   10
#define NS     84
#define NPTS   (R_RAYS*NS)      // 344064
#define HDIM   256
#define RAWS   56               // padded raw stride (54 valid)
#define ROWS   128              // rows (points) per workgroup in MLP
#define SUBT   (ROWS/16)        // 8 M-subtiles per block

// ---------------- workspace layout (bytes) ----------------
constexpr size_t WS_Z    = 0;                                        // float[NPTS]
constexpr size_t WS_FLAG = WS_Z + (size_t)NPTS*4;                    // u8[NPTS]
constexpr size_t WS_F1   = (WS_FLAG + (size_t)NPTS + 255) & ~(size_t)255; // half[2*16*512]
constexpr size_t WS_F2   = WS_F1 + (size_t)2*16*512*2;               // half[8*16*512]
constexpr size_t WS_F3   = WS_F2 + (size_t)8*16*512*2;               // half[8*4*512]
constexpr size_t WS_B1   = WS_F3 + (size_t)8*4*512*2;                // float[256]
constexpr size_t WS_B2   = WS_B1 + 256*4;                            // float[256]
constexpr size_t WS_B3   = WS_B2 + 256*4;                            // float[64]
constexpr size_t WS_RAW  = WS_B3 + 64*4;                             // float[NPTS*RAWS]

// ---------------- output layout (float elements) ----------------
constexpr int O_RGB = 0;
constexpr int O_DEP = O_RGB + R_RAYS*3;
constexpr int O_ACC = O_DEP + R_RAYS;
constexpr int O_SEM = O_ACC + R_RAYS;
constexpr int O_SFX = O_SEM + R_RAYS*50;
constexpr int O_W   = O_SFX + R_RAYS*50;
constexpr int O_OH  = O_W   + NPTS;

__device__ __forceinline__ float hash01(uint32_t x){
  x ^= x>>17; x *= 0xed5ad4bbu; x ^= x>>11; x *= 0xac4c1b51u;
  x ^= x>>15; x *= 0x31848babu; x ^= x>>14;
  return (float)(x>>8)*(1.0f/16777216.0f);
}

// =================== kernel 1: sampling / one-hot / flags ===================
__global__ void sample_kernel(const float* __restrict__ rays,
                              const float* __restrict__ inter,
                              float* __restrict__ zbuf,
                              uint8_t* __restrict__ flags,
                              float* __restrict__ oh_out){
  int r = blockIdx.x*blockDim.x + threadIdx.x;
  if (r >= R_RAYS) return;

  float nearb[NBOX], farb[NBOX]; int lab[NBOX];
  for (int n=0;n<NBOX;n++){
    const float* it = inter + ((size_t)r*NBOX+n)*4;
    nearb[n]=it[0]; farb[n]=it[1];
    int li=(int)it[3]; lab[n] = li<0?0:(li>49?49:li);
  }
  float zs[80], zb[20];
  for (int n=0;n<NBOX;n++){
    float nm=nearb[n], fm=farb[n];
    if (fm>150.f){ nm=-1.f; fm=-1.f; }
    for (int i=0;i<8;i++){
      float t=(float)i*(1.f/7.f);
      zs[n*8+i]= nm*(1.f-t)+fm*t;
    }
    zb[n]    = nm - 1e-5f;
    zb[10+n] = fm + 1e-5f;
  }
  for (int i=1;i<80;i++){ float k=zs[i]; int j=i-1;
    while(j>=0 && zs[j]>k){ zs[j+1]=zs[j]; j--; } zs[j+1]=k; }

  float z84[NS];
  // sample_pdf with uniform weights -> closed-form piecewise-linear CDF inverse
  for (int j=0;j<64;j++){
    float u  = (float)j*(1.f/63.f);
    float fi = u*79.f;
    int ib = (int)floorf(fi); if (ib<0) ib=0; if (ib>79) ib=79;
    int ia = ib+1; if (ia>79) ia=79;
    float cb = (float)ib*(1.f/79.f);
    float ca = (float)ia*(1.f/79.f);
    float den = (ca-cb < 1e-5f) ? 1.f : (ca-cb);
    float t = (u - cb)/den;
    z84[j] = zs[ib] + t*(zs[ia]-zs[ib]);
  }
  for (int n=0;n<20;n++) z84[64+n]=zb[n];
  for (int k=0;k<NS;k++)
    if (z84[k] < 0.f)
      z84[k] = 150.f + 20.f*hash01((uint32_t)(r*NS+k)*2654435761u + 12345u);
  for (int i=1;i<NS;i++){ float k=z84[i]; int j=i-1;
    while(j>=0 && z84[j]>k){ z84[j+1]=z84[j]; j--; } z84[j+1]=k; }
  for (int k=0;k<NS;k++) zbuf[(size_t)r*NS+k]=z84[k];

  const int MS[8]={39,27,28,29,30,31, 9,35};
  const int MD[8]={41,26,26,26,26,26, 8,13};
  for (int s=0;s<NS;s++){
    float zv=z84[s];
    unsigned long long oh=0ull; bool bound=false;
    for (int n=0;n<NBOX;n++){
      if (zv>nearb[n] && zv<farb[n]) oh |= 1ull<<lab[n];
      float df=zv-farb[n], dn=nearb[n]-zv;
      bound = bound || (df<0.001f && df>0.f) || (dn>0.f && dn<0.001f);
    }
    bool empty = (oh==0ull);
    bool bbox = empty && (zv<150.f);
    bool bg   = empty && (zv>150.f);
    if (bbox) oh |= 1ull;
    if (bg)   oh |= (1ull<<23);
    for (int m=0;m<8;m++)
      if ((oh>>MS[m])&1ull){ oh |= 1ull<<MD[m]; oh &= ~(1ull<<MS[m]); }
    int p = r*NS+s;
    flags[p] = (bbox || bound || (bg && s < NS-5)) ? 1 : 0;
    float* o = oh_out + (size_t)p*50;
    for (int c=0;c<50;c++) o[c] = ((oh>>c)&1ull)?1.f:0.f;
  }
}

// =================== kernel 2: weights -> WMMA B-fragment layout ===================
// frag index: ((kt*NT + nt)*32 + lane)*16 + h ; B layout: K = kt*32 + h + 16*laneHi, N = nt*16 + (lane&15)
__global__ void prep_kernel(const float* __restrict__ W1,const float* __restrict__ b1,
                            const float* __restrict__ W2,const float* __restrict__ b2,
                            const float* __restrict__ W3,const float* __restrict__ b3,
                            _Float16* __restrict__ f1,_Float16* __restrict__ f2,_Float16* __restrict__ f3,
                            float* __restrict__ bb1,float* __restrict__ bb2,float* __restrict__ bb3){
  int gid = blockIdx.x*blockDim.x+threadIdx.x;
  const int N1=2*16*512, N2=8*16*512, N3=8*4*512;
  if (gid < N1){
    int h=gid&15, lane=(gid>>4)&31, t=gid>>9; int nt=t&15, kt=t>>4;
    int k = kt*32 + h + ((lane>>4)<<4);
    int n = (nt<<4) + (lane&15);
    f1[gid] = (_Float16)(k<63 ? W1[k*256+n] : 0.f);
  } else if (gid < N1+N2){
    int idx=gid-N1; int h=idx&15, lane=(idx>>4)&31, t=idx>>9; int nt=t&15, kt=t>>4;
    int k = kt*32 + h + ((lane>>4)<<4);
    int n = (nt<<4) + (lane&15);
    f2[idx] = (_Float16)W2[k*256+n];
  } else if (gid < N1+N2+N3){
    int idx=gid-N1-N2; int h=idx&15, lane=(idx>>4)&31, t=idx>>9; int nt=t&3, kt=t>>2;
    int k = kt*32 + h + ((lane>>4)<<4);
    int n = (nt<<4) + (lane&15);
    f3[idx] = (_Float16)(n<54 ? W3[k*54+n] : 0.f);
  }
  if (gid < 256){ bb1[gid]=b1[gid]; bb2[gid]=b2[gid]; }
  if (gid < 64){ bb3[gid] = (gid<54) ? b3[gid] : 0.f; }
}

// =================== kernel 3: fused MLP via WMMA ===================
__device__ __forceinline__ v16h load_a_frag(const _Float16* src, int m16, int koff){
  const _Float16* ap = src + m16*HDIM + koff;   // halfs 0..7  -> K..K+7
  half8 a0 = *(const half8*)(ap);               // halfs 8..15 -> K+8..K+15
  half8 a1 = *(const half8*)(ap+8);
  return __builtin_shufflevector(a0,a1,0,1,2,3,4,5,6,7,8,9,10,11,12,13,14,15);
}

__global__ void __launch_bounds__(256)
mlp_kernel(const float* __restrict__ rays,
           const float* __restrict__ zbuf,
           const _Float16* __restrict__ f1,
           const _Float16* __restrict__ f2,
           const _Float16* __restrict__ f3,
           const float* __restrict__ bb1,
           const float* __restrict__ bb2,
           const float* __restrict__ bb3,
           float* __restrict__ raw){
  __shared__ __align__(32) _Float16 bufA[ROWS*HDIM];   // 64KB
  __shared__ __align__(32) _Float16 bufB[ROWS*HDIM];   // 64KB
  const int lane = threadIdx.x & 31;
  const int wv   = threadIdx.x >> 5;      // 0..7
  const int p0   = blockIdx.x*ROWS;

  // ---- stage positional-encoding features into bufA (128 rows, cols 0..63) ----
  {
    int frow = threadIdx.x >> 1;          // 0..127
    int cb   = (threadIdx.x & 1)*32;
    int p  = p0 + frow;
    int rr = p/NS;
    float zv = zbuf[p];
    const float* ray = rays + (size_t)rr*6;
    float dx=ray[3],dy=ray[4],dz=ray[5];
    float sc = sqrtf(dx*dx+dy*dy+dz*dz);
    float q = zv/sc;
    float px=ray[0]+dx*q, py=ray[1]+dy*q, pz=ray[2]+dz*q;
    for (int jj=0;jj<32;jj++){
      int k = cb+jj;
      float v;
      if (k<3)       v = (k==0)?px:((k==1)?py:pz);
      else if (k<63){
        int j=k-3; int f=j/6; int rm=j-6*f;
        float base = (rm==0||rm==3)?px:((rm==1||rm==4)?py:pz);
        float ang = base * (float)(1<<f);
        v = (rm<3) ? __sinf(ang) : __cosf(ang);
      } else v = 0.f;
      bufA[frow*HDIM + k] = (_Float16)v;
    }
  }
  __syncthreads();

  const int m16 = lane & 15;
  const int hi8 = (lane>>4)*8;
  const int kA  = (lane>>4)*16;    // A-fragment K base offset for this half-wave
  const v8f vzero = {0.f,0.f,0.f,0.f,0.f,0.f,0.f,0.f};

  // ---- layer 1: feat(64) x W1 -> 256, relu, into bufB. wave handles nt = wv, wv+8 ----
  for (int ni=0; ni<2; ni++){
    const int nt = wv + ni*8;
    v8f c[SUBT];
    #pragma unroll
    for (int mt=0;mt<SUBT;mt++) c[mt]=vzero;
    #pragma unroll
    for (int kt=0; kt<2; kt++){
      v16h b = *(const v16h*)(f1 + (size_t)((kt*16+nt)*32 + lane)*16);
      #pragma unroll
      for (int mt=0;mt<SUBT;mt++){
        v16h a = load_a_frag(bufA + mt*16*HDIM, m16, kt*32 + kA);
        c[mt] = __builtin_amdgcn_wmma_f32_16x16x32_f16(false,a,false,b,(short)0,c[mt],false,false);
      }
    }
    float bv = bb1[nt*16 + m16];
    #pragma unroll
    for (int mt=0;mt<SUBT;mt++){
      #pragma unroll
      for (int q=0;q<8;q++){
        float v = c[mt][q]+bv; v = v>0.f ? v : 0.f;
        bufB[(mt*16+hi8+q)*HDIM + nt*16 + m16] = (_Float16)v;
      }
    }
  }
  __syncthreads();

  // ---- layer 2: 256 x W2 -> 256, relu, into bufA ----
  for (int ni=0; ni<2; ni++){
    const int nt = wv + ni*8;
    v8f c[SUBT];
    #pragma unroll
    for (int mt=0;mt<SUBT;mt++) c[mt]=vzero;
    #pragma unroll
    for (int kt=0; kt<8; kt++){
      v16h b = *(const v16h*)(f2 + (size_t)((kt*16+nt)*32 + lane)*16);
      #pragma unroll
      for (int mt=0;mt<SUBT;mt++){
        v16h a = load_a_frag(bufB + mt*16*HDIM, m16, kt*32 + kA);
        c[mt] = __builtin_amdgcn_wmma_f32_16x16x32_f16(false,a,false,b,(short)0,c[mt],false,false);
      }
    }
    float bv = bb2[nt*16 + m16];
    #pragma unroll
    for (int mt=0;mt<SUBT;mt++){
      #pragma unroll
      for (int q=0;q<8;q++){
        float v = c[mt][q]+bv; v = v>0.f ? v : 0.f;
        bufA[(mt*16+hi8+q)*HDIM + nt*16 + m16] = (_Float16)v;
      }
    }
  }
  __syncthreads();

  // ---- layer 3: 256 x W3 -> 54 (padded 64), f32 raw. waves 0..3 only (uniform branch) ----
  if (wv < 4){
    const int nt = wv;
    v8f c[SUBT];
    #pragma unroll
    for (int mt=0;mt<SUBT;mt++) c[mt]=vzero;
    #pragma unroll
    for (int kt=0; kt<8; kt++){
      v16h b = *(const v16h*)(f3 + (size_t)((kt*4+nt)*32 + lane)*16);
      #pragma unroll
      for (int mt=0;mt<SUBT;mt++){
        v16h a = load_a_frag(bufA + mt*16*HDIM, m16, kt*32 + kA);
        c[mt] = __builtin_amdgcn_wmma_f32_16x16x32_f16(false,a,false,b,(short)0,c[mt],false,false);
      }
    }
    float bv = bb3[nt*16 + m16];
    int col = nt*16 + m16;
    if (col < RAWS){
      #pragma unroll
      for (int mt=0;mt<SUBT;mt++){
        #pragma unroll
        for (int q=0;q<8;q++)
          raw[(size_t)(p0 + mt*16 + hi8 + q)*RAWS + col] = c[mt][q]+bv;
      }
    }
  }
}

// =================== kernel 4: volume rendering (weights/rgb/depth/acc) ===================
__global__ void render_kernel(const float* __restrict__ rays,
                              const float* __restrict__ zbuf,
                              const uint8_t* __restrict__ flags,
                              const float* __restrict__ raw,
                              float* __restrict__ out){
  int r = blockIdx.x*blockDim.x+threadIdx.x;
  if (r >= R_RAYS) return;
  const float* ray = rays + (size_t)r*6;
  float dx=ray[3],dy=ray[4],dz=ray[5];
  float sc = sqrtf(dx*dx+dy*dy+dz*dz);
  float inv = 1.f/sc;
  float T=1.f, cr=0.f,cg=0.f,cb=0.f, dep=0.f, acc=0.f;
  for (int s=0;s<NS;s++){
    int p = r*NS+s;
    float z = zbuf[p];
    float dist = (s<NS-1) ? (zbuf[p+1]-z) : 1e10f*sc;   // (z_n diff)*scale == z diff
    float sg = flags[p] ? 0.f : raw[(size_t)p*RAWS+3];
    sg = sg>0.f ? sg : 0.f;
    float alpha = 1.f - __expf(-sg*dist);
    float w = alpha*T;
    T *= (1.f - alpha + 1e-10f);
    out[O_W + p] = w;
    float r0 = raw[(size_t)p*RAWS+0];
    float r1 = raw[(size_t)p*RAWS+1];
    float r2 = raw[(size_t)p*RAWS+2];
    cr += w/(1.f+__expf(-r0));
    cg += w/(1.f+__expf(-r1));
    cb += w/(1.f+__expf(-r2));
    dep += w*z*inv;
    acc += w;
  }
  out[O_RGB + r*3+0]=cr;
  out[O_RGB + r*3+1]=cg;
  out[O_RGB + r*3+2]=cb;
  out[O_DEP + r]=dep;
  out[O_ACC + r]=acc;
}

// =================== kernel 5: semantic maps ===================
__global__ void sem_kernel(const float* __restrict__ raw,
                           float* __restrict__ out){
  int r = blockIdx.x;
  int c = threadIdx.x;
  if (c >= 50) return;
  float sm=0.f, sf=0.f;
  for (int s=0;s<NS;s++){
    int p = r*NS+s;
    float w = out[O_W+p];
    sm += w*raw[(size_t)p*RAWS+4+c];
    sf += w*out[O_OH + (size_t)p*50 + c];
  }
  out[O_SEM + r*50 + c] = sm;
  out[O_SFX + r*50 + c] = sf;
}

// =================== launch ===================
extern "C" void kernel_launch(void* const* d_in, const int* in_sizes, int n_in,
                              void* d_out, int out_size, void* d_ws, size_t ws_size,
                              hipStream_t stream){
  const float* rays  = (const float*)d_in[0];
  const float* inter = (const float*)d_in[1];
  const float* W1 = (const float*)d_in[2];
  const float* b1 = (const float*)d_in[3];
  const float* W2 = (const float*)d_in[4];
  const float* b2 = (const float*)d_in[5];
  const float* W3 = (const float*)d_in[6];
  const float* b3 = (const float*)d_in[7];
  float* out = (float*)d_out;
  char*  ws  = (char*)d_ws;

  float*    zbuf  = (float*)(ws + WS_Z);
  uint8_t*  flags = (uint8_t*)(ws + WS_FLAG);
  _Float16* f1    = (_Float16*)(ws + WS_F1);
  _Float16* f2    = (_Float16*)(ws + WS_F2);
  _Float16* f3    = (_Float16*)(ws + WS_F3);
  float*    bb1   = (float*)(ws + WS_B1);
  float*    bb2   = (float*)(ws + WS_B2);
  float*    bb3   = (float*)(ws + WS_B3);
  float*    raw   = (float*)(ws + WS_RAW);

  prep_kernel  <<<384, 256, 0, stream>>>(W1,b1,W2,b2,W3,b3, f1,f2,f3, bb1,bb2,bb3);
  sample_kernel<<<32, 128, 0, stream>>>(rays, inter, zbuf, flags, out + O_OH);
  mlp_kernel   <<<NPTS/ROWS, 256, 0, stream>>>(rays, zbuf, f1,f2,f3, bb1,bb2,bb3, raw);
  render_kernel<<<16, 256, 0, stream>>>(rays, zbuf, flags, raw, out);
  sem_kernel   <<<R_RAYS, 64, 0, stream>>>(raw, out);
}
